// PipelinedMoEBlock_42193758716264
// MI455X (gfx1250) — matrix operand order repaired and checked by reading, not compiled, not run on hardware
//
#include <hip/hip_runtime.h>

typedef __attribute__((ext_vector_type(16))) _Float16 v16h;
typedef __attribute__((ext_vector_type(8)))  _Float16 v8h;
typedef __attribute__((ext_vector_type(8)))  float    v8f;
typedef __attribute__((ext_vector_type(4)))  int      i32x4;
typedef __attribute__((address_space(1))) i32x4 gi4;   // global int4
typedef __attribute__((address_space(3))) i32x4 li4;   // LDS int4

#define NTOK   2048
#define DMODEL 1024
#define NHEAD  16
#define DHEAD  64
#define TSEQ   512
#define NBATCH 4
#define HIDDEN 4096
#define NEXP   8

// ---------------- CDNA5 async global->LDS copy (16B per lane, ASYNCcnt) ----
__device__ __forceinline__ void cp_async16(void* lds, const void* g) {
#if __has_builtin(__builtin_amdgcn_global_load_async_to_lds_b128)
  __builtin_amdgcn_global_load_async_to_lds_b128(
      (gi4*)(void*)g, (li4*)lds, 0, 0);
#else
  unsigned l32 = (unsigned)(unsigned long long)(li4*)lds;
  asm volatile("global_load_async_to_lds_b128 %0, %1, off"
               :: "v"(l32), "v"((gi4*)(void*)g)
               : "memory");
#endif
}
template <int N>
__device__ __forceinline__ void wait_asynccnt() {
#if __has_builtin(__builtin_amdgcn_s_wait_asynccnt)
  __builtin_amdgcn_s_wait_asynccnt((short)N);
#else
  asm volatile("s_wait_asynccnt %0" :: "i"(N) : "memory");
#endif
}

// ---------------------------------------------------------------- LayerNorm
__global__ __launch_bounds__(256) void k_ln(const float* __restrict__ x,
                                            const float* __restrict__ g,
                                            const float* __restrict__ b,
                                            _Float16* __restrict__ out) {
  const int lane = threadIdx.x & 31;
  const int wave = threadIdx.x >> 5;
  const int tok  = blockIdx.x * 8 + wave;
  const float* xr = x + (size_t)tok * DMODEL;
  float4 v[8];
  float s = 0.f;
  for (int j = 0; j < 8; ++j) {
    v[j] = *(const float4*)(xr + j * 128 + lane * 4);
    s += v[j].x + v[j].y + v[j].z + v[j].w;
  }
  for (int o = 16; o > 0; o >>= 1) s += __shfl_xor(s, o, 32);
  const float mu = s * (1.0f / DMODEL);
  float q = 0.f;
  for (int j = 0; j < 8; ++j) {
    float a0 = v[j].x - mu, a1 = v[j].y - mu, a2 = v[j].z - mu, a3 = v[j].w - mu;
    q += a0 * a0 + a1 * a1 + a2 * a2 + a3 * a3;
  }
  for (int o = 16; o > 0; o >>= 1) q += __shfl_xor(q, o, 32);
  const float rs = rsqrtf(q * (1.0f / DMODEL) + 1e-5f);
  _Float16* orow = out + (size_t)tok * DMODEL;
  for (int j = 0; j < 8; ++j) {
    const int d = j * 128 + lane * 4;
    const float4 gw = *(const float4*)(g + d);
    const float4 bw = *(const float4*)(b + d);
    orow[d + 0] = (_Float16)((v[j].x - mu) * rs * gw.x + bw.x);
    orow[d + 1] = (_Float16)((v[j].y - mu) * rs * gw.y + bw.y);
    orow[d + 2] = (_Float16)((v[j].z - mu) * rs * gw.z + bw.z);
    orow[d + 3] = (_Float16)((v[j].w - mu) * rs * gw.w + bw.w);
  }
}

// ---------------- weight transpose-convert: fp32 [K,N] -> f16 [N,K] --------
__global__ __launch_bounds__(256) void k_trans(const float* __restrict__ src,
                                               _Float16* __restrict__ dst,
                                               int K, int N) {
  __shared__ float t[32][33];
  const int k0 = blockIdx.y * 32, n0 = blockIdx.x * 32;
  const int tx = threadIdx.x & 31, ty = threadIdx.x >> 5;
  for (int i = 0; i < 4; ++i) {
    const int r = ty + i * 8;
    t[r][tx] = src[(size_t)(k0 + r) * N + n0 + tx];
  }
  __syncthreads();
  for (int i = 0; i < 4; ++i) {
    const int r = ty + i * 8;
    dst[(size_t)(n0 + r) * K + k0 + tx] = (_Float16)t[tx][r];
  }
}

// ----------------------- split QKV: per-head f16; V stored transposed ------
__global__ __launch_bounds__(256) void k_split(const float* __restrict__ qkv,
                                               _Float16* __restrict__ qh,
                                               _Float16* __restrict__ kh,
                                               _Float16* __restrict__ vhT) {
  const size_t i   = (size_t)blockIdx.x * 256 + threadIdx.x;  // < NTOK*3*DMODEL
  const int    col = (int)(i % (3 * DMODEL));
  const size_t tok = i / (3 * DMODEL);
  const int    bb  = (int)(tok / TSEQ), tt = (int)(tok % TSEQ);
  const int    which = col / DMODEL;
  const int    dp  = col % DMODEL;
  const int    hh  = dp / DHEAD, dd = dp % DHEAD;
  const float  val = qkv[i];
  if (which == 2) {  // V transposed: [b,h,dh,t] so P*V's B operand is [N,K]
    vhT[(((size_t)bb * NHEAD + hh) * DHEAD + dd) * TSEQ + tt] = (_Float16)val;
  } else {
    const size_t o = (((size_t)bb * NHEAD + hh) * TSEQ + tt) * DHEAD + dd;
    if (which == 0) qh[o] = (_Float16)(val * 0.125f);   // fold 1/sqrt(dh)
    else            kh[o] = (_Float16)val;
  }
}

// ------------------------------------------------- in-place row softmax (f16)
__global__ __launch_bounds__(256) void k_softmax(_Float16* __restrict__ S) {
  const int lane = threadIdx.x & 31;
  const int wave = threadIdx.x >> 5;
  const size_t row = (size_t)blockIdx.x * 8 + wave;
  _Float16* r = S + row * (size_t)TSEQ;
  v8h c0 = *(v8h*)(r + lane * 16);
  v8h c1 = *(v8h*)(r + lane * 16 + 8);
  float f[16];
  for (int q = 0; q < 8; ++q) { f[q] = (float)c0[q]; f[q + 8] = (float)c1[q]; }
  float m = -1e30f;
  for (int q = 0; q < 16; ++q) m = fmaxf(m, f[q]);
  for (int o = 16; o > 0; o >>= 1) m = fmaxf(m, __shfl_xor(m, o, 32));
  float s = 0.f;
  for (int q = 0; q < 16; ++q) { f[q] = expf(f[q] - m); s += f[q]; }
  for (int o = 16; o > 0; o >>= 1) s += __shfl_xor(s, o, 32);
  const float inv = 1.0f / s;
  v8h o0, o1;
  for (int q = 0; q < 8; ++q) { o0[q] = (_Float16)(f[q] * inv); o1[q] = (_Float16)(f[q + 8] * inv); }
  *(v8h*)(r + lane * 16)     = o0;
  *(v8h*)(r + lane * 16 + 8) = o1;
}

// -------------------------------------- permute heads back: [b,h,t,dh]->[tok,D]
__global__ __launch_bounds__(256) void k_permute(const _Float16* __restrict__ ctx,
                                                 _Float16* __restrict__ out) {
  const size_t i   = (size_t)blockIdx.x * 256 + threadIdx.x;  // < NTOK*DMODEL
  const int    d   = (int)(i % DMODEL);
  const size_t tok = i / DMODEL;
  const int    bb  = (int)(tok / TSEQ), tt = (int)(tok % TSEQ);
  const int    hh  = d / DHEAD, dd = d % DHEAD;
  out[i] = ctx[(((size_t)bb * NHEAD + hh) * TSEQ + tt) * DHEAD + dd];
}

// ---------------------------------------- gate: softmax + top-2 + renormalize
__global__ __launch_bounds__(256) void k_gate(const _Float16* __restrict__ t,
                                              const float* __restrict__ wg,
                                              float* __restrict__ wcomb) {
  const int lane = threadIdx.x & 31;
  const int wave = threadIdx.x >> 5;
  const int tok  = blockIdx.x * 8 + wave;
  const _Float16* tr = t + (size_t)tok * DMODEL;
  float a0=0,a1=0,a2=0,a3=0,a4=0,a5=0,a6=0,a7=0;
  for (int d = lane; d < DMODEL; d += 32) {
    const float tv = (float)tr[d];
    const float4 w0 = *(const float4*)(wg + (size_t)d * 8);
    const float4 w1 = *(const float4*)(wg + (size_t)d * 8 + 4);
    a0 += tv * w0.x; a1 += tv * w0.y; a2 += tv * w0.z; a3 += tv * w0.w;
    a4 += tv * w1.x; a5 += tv * w1.y; a6 += tv * w1.z; a7 += tv * w1.w;
  }
  float acc[8] = {a0,a1,a2,a3,a4,a5,a6,a7};
  for (int e = 0; e < 8; ++e)
    for (int o = 16; o > 0; o >>= 1) acc[e] += __shfl_xor(acc[e], o, 32);
  if (lane == 0) {
    float m = acc[0];
    for (int e = 1; e < 8; ++e) m = fmaxf(m, acc[e]);
    float p[8], s = 0.f;
    for (int e = 0; e < 8; ++e) { p[e] = expf(acc[e] - m); s += p[e]; }
    for (int e = 0; e < 8; ++e) p[e] /= s;
    int i1 = 0;
    for (int e = 1; e < 8; ++e) if (p[e] > p[i1]) i1 = e;
    int i2 = (i1 == 0) ? 1 : 0;
    for (int e = 0; e < 8; ++e) if (e != i1 && p[e] > p[i2]) i2 = e;
    const float z = p[i1] + p[i2];
    float* o = wcomb + (size_t)tok * 8;
    for (int e = 0; e < 8; ++e)
      o[e] = (e == i1) ? p[i1] / z : (e == i2) ? p[i2] / z : 0.f;
  }
}

// ------------------- async double-buffered WMMA GEMM:  A f16[M,K], B f16[N,K]
// EPI: 0 f32=acc+bias | 1 f16=acc | 2 f32=resid+acc+bias | 3 f16=gelu(acc+bias)
//      4 f32=resid+roww*(acc+bias) | 5 f32+=roww*(acc+bias)
template <int BM, int BN, int EPI>
__global__ __launch_bounds__(256) void k_gemm(
    const _Float16* __restrict__ A, long long sAz, int lda,
    const _Float16* __restrict__ Bt, long long sBz, int ldb,
    const float* __restrict__ bias,
    float* __restrict__ outF, _Float16* __restrict__ outH, long long sOz, int ldo,
    const float* __restrict__ resid, const float* __restrict__ roww, int rowws,
    int K) {
  constexpr int BK = 32;
  constexpr int LS = BK + 8;                 // padded LDS row (halves)
  constexpr int WM = BM / 4, WN = BN / 2;    // 4x2 wave grid (8 waves)
  constexpr int TM = WM / 16, TN = WN / 16;
  constexpr int NB = BN / 64;                // B async chunks per wave (2 or 1)
  constexpr int NCH = 2 + NB;                // async issues per wave per step
  __shared__ _Float16 smem[2][(BM + BN) * LS];

  const int tid = threadIdx.x;
  const int lane = tid & 31, wid = tid >> 5;
  const int wm = wid & 3, wn = wid >> 2;
  const int z  = blockIdx.z;
  const int m0 = blockIdx.y * BM, n0 = blockIdx.x * BN;
  A  += (size_t)z * sAz;
  Bt += (size_t)z * sBz;

  // per-lane async copy addressing: chunk = 8 rows x 64B; lane covers
  // row = chunk*8 + lane/4, 16B piece (lane%4)*8 halves.
  const int r8 = lane >> 2;
  const int s8 = (lane & 3) * 8;
  const _Float16* ag[2];
  const _Float16* bg[NB];
  int arow[2], brow[NB];
  for (int c = 0; c < 2; ++c) {
    arow[c] = (wid * 2 + c) * 8 + r8;
    ag[c] = A + (size_t)(m0 + arow[c]) * lda + s8;
  }
  for (int c = 0; c < NB; ++c) {
    brow[c] = (wid * NB + c) * 8 + r8;
    bg[c] = Bt + (size_t)(n0 + brow[c]) * ldb + s8;
  }

  v8f acc[TM][TN];
  const v8f vzero = {0.f, 0.f, 0.f, 0.f, 0.f, 0.f, 0.f, 0.f};
  for (int i = 0; i < TM; ++i)
    for (int j = 0; j < TN; ++j) acc[i][j] = vzero;

  auto issue = [&](int buf) {
    _Float16* ab = &smem[buf][0];
    _Float16* bb = &smem[buf][BM * LS];
    for (int c = 0; c < 2; ++c) { cp_async16(ab + arow[c] * LS + s8, ag[c]); ag[c] += BK; }
    for (int c = 0; c < NB; ++c) { cp_async16(bb + brow[c] * LS + s8, bg[c]); bg[c] += BK; }
  };

  const int nsteps = K / BK;
  issue(0);
  for (int s = 0; s < nsteps; ++s) {
    const int cur = s & 1;
    if (s + 1 < nsteps) { issue(cur ^ 1); wait_asynccnt<NCH>(); }
    else                { wait_asynccnt<0>(); }
    __syncthreads();                       // all waves' tiles landed

    const _Float16* sA = &smem[cur][0];
    const _Float16* sB = &smem[cur][BM * LS];
    v16h af[TM], bfr[TN];
    const int r16 = lane & 15;
    const int ka  = (lane < 16) ? 0 : 8;   // A: lanes 0-15 K0..7/16..23, 16-31 K8..15/24..31
    for (int i = 0; i < TM; ++i) {
      const _Float16* p = sA + (wm * WM + i * 16 + r16) * LS + ka;
      const v8h lo = *(const v8h*)p;
      const v8h hi = *(const v8h*)(p + 16);
      v16h t;
      for (int q = 0; q < 8; ++q) { t[q] = lo[q]; t[q + 8] = hi[q]; }
      af[i] = t;
    }
    const int kb = (lane < 16) ? 0 : 8;    // B: lanes 0-15 K0..7/16..23, 16-31 K8..15/24..31
    for (int j = 0; j < TN; ++j) {
      const _Float16* p = sB + (wn * WN + j * 16 + r16) * LS + kb;
      const v8h lo = *(const v8h*)p;
      const v8h hi = *(const v8h*)(p + 16);
      v16h t;
      for (int q = 0; q < 8; ++q) { t[q] = lo[q]; t[q + 8] = hi[q]; }
      bfr[j] = t;
    }
    for (int i = 0; i < TM; ++i)
      for (int j = 0; j < TN; ++j)
        acc[i][j] = __builtin_amdgcn_wmma_f32_16x16x32_f16(
            false, af[i], false, bfr[j], (short)0, acc[i][j], false, false);
    __syncthreads();                       // protect buf before next overwrite
  }

  // ---- epilogue (C layout: VGPR r -> row r (+8 lanes 16-31), lane&15 -> col)
  float*    oF = outF ? outF + (size_t)z * sOz : nullptr;
  _Float16* oH = outH ? outH + (size_t)z * sOz : nullptr;
  const int rofs = (lane >> 4) << 3;
  for (int i = 0; i < TM; ++i)
    for (int j = 0; j < TN; ++j)
      for (int rr = 0; rr < 8; ++rr) {
        const int row = m0 + wm * WM + i * 16 + rr + rofs;
        const int col = n0 + wn * WN + j * 16 + (lane & 15);
        const float v = acc[i][j][rr];
        const size_t oi = (size_t)row * ldo + col;
        if (EPI == 0)      oF[oi] = v + bias[col];
        else if (EPI == 1) oH[oi] = (_Float16)v;
        else if (EPI == 2) oF[oi] = resid[oi] + v + bias[col];
        else if (EPI == 3) {
          const float u = v + bias[col];
          const float g = 0.5f * u * (1.0f + tanhf(0.7978845608028654f * (u + 0.044715f * u * u * u)));
          oH[oi] = (_Float16)g;
        } else if (EPI == 4) oF[oi] = resid[oi] + roww[(size_t)row * rowws] * (v + bias[col]);
        else               oF[oi] += roww[(size_t)row * rowws] * (v + bias[col]);
      }
}

// --------------------------------------------------------------- launcher
extern "C" void kernel_launch(void* const* d_in, const int* in_sizes, int n_in,
                              void* d_out, int out_size, void* d_ws, size_t ws_size,
                              hipStream_t stream) {
  const float* x      = (const float*)d_in[0];
  const float* ln1_w  = (const float*)d_in[1];
  const float* ln1_b  = (const float*)d_in[2];
  const float* ln2_w  = (const float*)d_in[3];
  const float* ln2_b  = (const float*)d_in[4];
  const float* w_qkv  = (const float*)d_in[5];
  const float* b_qkv  = (const float*)d_in[6];
  const float* w_o    = (const float*)d_in[7];
  const float* b_o    = (const float*)d_in[8];
  const float* w_gate = (const float*)d_in[9];
  const float* w1     = (const float*)d_in[10];
  const float* b1     = (const float*)d_in[11];
  const float* w2     = (const float*)d_in[12];
  const float* b2     = (const float*)d_in[13];
  float* out = (float*)d_out;

  // workspace arena (byte offsets, dead-region aliasing)
  char* ws = (char*)d_ws;
  _Float16* xh    = (_Float16*)(ws + 0);               // dead after QKV
  _Float16* qh    = (_Float16*)(ws + 4194304);
  _Float16* kh    = (_Float16*)(ws + 8388608);
  _Float16* vhT   = (_Float16*)(ws + 12582912);        // [b,h,dh,t]
  float*    qkv   = (float*)   (ws + 16777216);        // aliases S (dead first)
  _Float16* S     = (_Float16*)(ws + 16777216);        // in-place softmax -> P
  _Float16* ctxt  = (_Float16*)(ws + 50331648);        // [b,h,t,dh] f16
  _Float16* ctxh  = (_Float16*)(ws + 0);               // aliases xh
  float*    h     = (float*)   (ws + 54525952);        // residual stream
  _Float16* h2h   = (_Float16*)(ws + 62914560);        // LN2 out f16
  float*    wcomb = (float*)   (ws + 67108864);        // [N,8] combine weights
  _Float16* h1h   = (_Float16*)(ws + 67174400);        // [N,4096] gelu acts
  _Float16* wqkvT = (_Float16*)(ws + 83951616);        // [3D, D] f16
  _Float16* woT   = (_Float16*)(ws + 90243072);        // [D, D]  f16
  _Float16* w1Tb  = (_Float16*)(ws + 92340224);        // per-expert [H, D]
  _Float16* w2Tb  = (_Float16*)(ws + 100728832);       // per-expert [D, H]

  const dim3 blk(256);
  const long long SQ = (long long)TSEQ * DHEAD;
  const long long SS = (long long)TSEQ * TSEQ;

  // weight transpose-converts for attention projections
  k_trans<<<dim3(3 * DMODEL / 32, DMODEL / 32, 1), blk, 0, stream>>>(w_qkv, wqkvT, DMODEL, 3 * DMODEL);
  k_trans<<<dim3(DMODEL / 32, DMODEL / 32, 1), blk, 0, stream>>>(w_o, woT, DMODEL, DMODEL);

  // 1) LN1 -> f16
  k_ln<<<NTOK / 8, blk, 0, stream>>>(x, ln1_w, ln1_b, xh);
  // 2) QKV GEMM
  k_gemm<128, 128, 0><<<dim3(3 * DMODEL / 128, NTOK / 128, 1), blk, 0, stream>>>(
      xh, 0, DMODEL, wqkvT, 0, DMODEL, b_qkv,
      qkv, nullptr, 0, 3 * DMODEL, nullptr, nullptr, 0, DMODEL);
  // 3) split to per-head f16 (q scaled, V transposed)
  k_split<<<NTOK * 3 * DMODEL / 256, blk, 0, stream>>>(qkv, qh, kh, vhT);
  // 4) S = Q * K^T (batched over b*h)
  k_gemm<128, 128, 1><<<dim3(TSEQ / 128, TSEQ / 128, NBATCH * NHEAD), blk, 0, stream>>>(
      qh, SQ, DHEAD, kh, SQ, DHEAD, nullptr,
      nullptr, S, SS, TSEQ, nullptr, nullptr, 0, DHEAD);
  // 5) softmax rows in place
  k_softmax<<<NBATCH * NHEAD * TSEQ / 8, blk, 0, stream>>>(S);
  // 6) ctx = P * V  (B operand = V^T, [dh, T])
  k_gemm<128, 64, 1><<<dim3(DHEAD / 64, TSEQ / 128, NBATCH * NHEAD), blk, 0, stream>>>(
      S, SS, TSEQ, vhT, SQ, TSEQ, nullptr,
      nullptr, ctxt, SQ, DHEAD, nullptr, nullptr, 0, TSEQ);
  // 7) permute heads back
  k_permute<<<NTOK * DMODEL / 256, blk, 0, stream>>>(ctxt, ctxh);
  // 8) out-proj + residual: h = x + ctx*w_o + b_o
  k_gemm<128, 128, 2><<<dim3(DMODEL / 128, NTOK / 128, 1), blk, 0, stream>>>(
      ctxh, 0, DMODEL, woT, 0, DMODEL, b_o,
      h, nullptr, 0, DMODEL, x, nullptr, 0, DMODEL);
  // 9) LN2 -> f16
  k_ln<<<NTOK / 8, blk, 0, stream>>>(h, ln2_w, ln2_b, h2h);
  // 10) gate
  k_gate<<<NTOK / 8, blk, 0, stream>>>(h2h, w_gate, wcomb);
  // 11) MoE experts (lazy per-expert weight convert; d_out = h + sum_e w_e*FFN_e)
  for (int e = 0; e < NEXP; ++e) {
    k_trans<<<dim3(HIDDEN / 32, DMODEL / 32, 1), blk, 0, stream>>>(
        w1 + (size_t)e * DMODEL * HIDDEN, w1Tb, DMODEL, HIDDEN);
    k_trans<<<dim3(DMODEL / 32, HIDDEN / 32, 1), blk, 0, stream>>>(
        w2 + (size_t)e * HIDDEN * DMODEL, w2Tb, HIDDEN, DMODEL);
    k_gemm<128, 128, 3><<<dim3(HIDDEN / 128, NTOK / 128, 1), blk, 0, stream>>>(
        h2h, 0, DMODEL, w1Tb, 0, DMODEL, b1 + (size_t)e * HIDDEN,
        nullptr, h1h, 0, HIDDEN, nullptr, nullptr, 0, DMODEL);
    if (e == 0)
      k_gemm<128, 128, 4><<<dim3(DMODEL / 128, NTOK / 128, 1), blk, 0, stream>>>(
          h1h, 0, HIDDEN, w2Tb, 0, HIDDEN, b2 + (size_t)e * DMODEL,
          out, nullptr, 0, DMODEL, h, wcomb + e, NEXP, HIDDEN);
    else
      k_gemm<128, 128, 5><<<dim3(DMODEL / 128, NTOK / 128, 1), blk, 0, stream>>>(
          h1h, 0, HIDDEN, w2Tb, 0, HIDDEN, b2 + (size_t)e * DMODEL,
          out, nullptr, 0, DMODEL, h, wcomb + e, NEXP, HIDDEN);
  }
  (void)in_sizes; (void)n_in; (void)out_size; (void)ws_size;
}
